// TriangleMultiplication_64493228917018
// MI455X (gfx1250) — matrix operand order repaired and checked by reading, not compiled, not run on hardware
//
#include <hip/hip_runtime.h>
#include <hip/hip_bf16.h>

// Triangle multiplication (outgoing) for MI455X / gfx1250.
// bf16 WMMA (v_wmma_f32_16x16x32_bf16) + fp32 accumulation for all GEMM
// stages; bf16 intermediates to halve HBM traffic (23.3 TB/s roofline is
// balanced against bf16 matrix throughput at ~86 GFLOP total).
// The dominant triangle-contraction kernel double-buffers its LDS slabs with
// CDNA5 async-to-LDS copies (GLOBAL_LOAD_ASYNC_TO_LDS_B128 / ASYNCcnt) so
// HBM streaming overlaps the WMMA pipeline.
//
// Workspace layout (needs ~257 MiB):
//   [0, 192KiB)          wcat   : 6 x 128x128 bf16 weights (L,R,LG,RG,G,OUT)
//   [1MiB, +64MiB)       leftT  : bf16 [c][i][k]   (A-side of triangle GEMM)
//   [+64MiB, +64MiB)     rightT : bf16 [c][k][j]   (B-side, pre-transposed)
//   [+64MiB, +64MiB)     gateO  : bf16 [i][j][c]   sigmoid(inp@w_gate+b)
//   [+64MiB, +64MiB)     actB   : bf16 [c][i][j]   triangle contraction result

#define NN 512
#define CP 128
#define EPSV 1e-5f

typedef __attribute__((ext_vector_type(16))) __bf16 bf16x16;
typedef __attribute__((ext_vector_type(8)))  __bf16 bf16x8;
typedef __attribute__((ext_vector_type(8)))  float  f32x8;

union ABfrag { bf16x16 v; bf16x8 h[2]; };

__device__ __forceinline__ float sigm(float x) { return 1.0f / (1.0f + __expf(-x)); }

// CDNA5 async copy: 16B per lane, global -> LDS, tracked by ASYNCcnt.
// LDS destination address = low 32 bits of the flat pointer (addrspace(3)
// offset per the LDS aperture rule, ISA 10.2).
__device__ __forceinline__ void async_b128(unsigned lds_addr, const __bf16* g) {
  asm volatile("global_load_async_to_lds_b128 %0, %1, off"
               :: "v"(lds_addr), "v"((unsigned long long)(size_t)g)
               : "memory");
}

// ---------------------------------------------------------------- k_convert
__global__ __launch_bounds__(256) void k_convert(
    const float* __restrict__ w0, const float* __restrict__ w1,
    const float* __restrict__ w2, const float* __restrict__ w3,
    const float* __restrict__ w4, const float* __restrict__ w5,
    __bf16* __restrict__ wcat) {
  int idx = blockIdx.x * 256 + threadIdx.x;          // 6*16384 = 98304
  if (idx >= 6 * 16384) return;
  int p = idx >> 14;
  const float* src = (p == 0) ? w0 : (p == 1) ? w1 : (p == 2) ? w2
                    : (p == 3) ? w3 : (p == 4) ? w4 : w5;
  wcat[idx] = (__bf16)src[idx & 16383];
}

// ---------------------------------------------------------------- k_proj
// Block: 256 thr (8 waves). Super-tile: 8 'a' rows x 16 'b' cols of pairs.
// Per row-tile (fixed a, 16 b-rows): LN -> lnbuf bf16, then each wave computes
// its 16-output-channel slice for all 5 projections (left,right,lgate,rgate,gate).
__global__ __launch_bounds__(256) void k_proj(
    const float* __restrict__ pact, const float* __restrict__ pmask,
    const float* __restrict__ lnw, const float* __restrict__ lnb,
    const float* __restrict__ bL, const float* __restrict__ bR,
    const float* __restrict__ bLG, const float* __restrict__ bRG,
    const float* __restrict__ bG,
    const __bf16* __restrict__ wcat,
    __bf16* __restrict__ leftT, __bf16* __restrict__ rightT,
    __bf16* __restrict__ gateO) {
  __shared__ __bf16 lnbuf[16 * 136];    // LN'd rows, pitch 136 (16B aligned, bank-spread)
  __shared__ __bf16 rstage[128 * 136];  // right transpose stage: [c][b_local*8 + a_local]
  __shared__ __bf16 gstage[16 * 128];   // gate stage: [b_local][c]

  const int t = threadIdx.x;
  const int lane = t & 31;
  const int w = t >> 5;
  const int half = lane >> 4;
  const int ln16 = lane & 15;

  const int b0 = blockIdx.x * 16;
  const int a0 = blockIdx.y * 8;

  const int c0 = w * 16;
  const int c = c0 + ln16;

  const float vbL = bL[c], vbR = bR[c], vbLG = bLG[c], vbRG = bRG[c], vbG = bG[c];

  for (int rt = 0; rt < 8; ++rt) {
    const int a = a0 + rt;
    __syncthreads();
    // ---- LayerNorm: 16 rows (b) x 128 ch, 16 lanes per row, 8 ch per lane
    {
      const int r = t >> 4;
      const int s = t & 15;
      const int ch = s * 8;
      const float* src = pact + ((size_t)a * NN + (b0 + r)) * CP + ch;
      float x[8];
      float sum = 0.f, sq = 0.f;
#pragma unroll
      for (int i = 0; i < 8; ++i) { x[i] = src[i]; sum += x[i]; sq += x[i] * x[i]; }
#pragma unroll
      for (int off = 1; off < 16; off <<= 1) {
        sum += __shfl_xor(sum, off, 32);
        sq  += __shfl_xor(sq,  off, 32);
      }
      const float mean = sum * (1.0f / CP);
      const float var  = sq * (1.0f / CP) - mean * mean;
      const float rs   = rsqrtf(var + EPSV);
      bf16x8 y;
#pragma unroll
      for (int i = 0; i < 8; ++i)
        y[i] = (__bf16)((x[i] - mean) * rs * lnw[ch + i] + lnb[ch + i]);
      *(bf16x8*)&lnbuf[r * 136 + ch] = y;
    }
    __syncthreads();

    // ---- 5 projections, K=128, wave's channels c0..c0+15; A software-pipelined
    f32x8 acc[5];
#pragma unroll
    for (int p = 0; p < 5; ++p) acc[p] = (f32x8)(0.0f);
    const int kcb = half ? 8 : 0;
    ABfrag A;
    A.h[0] = *(const bf16x8*)&lnbuf[ln16 * 136 + kcb];
    A.h[1] = *(const bf16x8*)&lnbuf[ln16 * 136 + kcb + 16];
#pragma unroll
    for (int ks = 0; ks < 4; ++ks) {
      const int k0 = ks * 32;
      ABfrag An;
      if (ks < 3) {  // prefetch next A fragment before the WMMA group
        An.h[0] = *(const bf16x8*)&lnbuf[ln16 * 136 + k0 + 32 + kcb];
        An.h[1] = *(const bf16x8*)&lnbuf[ln16 * 136 + k0 + 32 + kcb + 16];
      }
#pragma unroll
      for (int p = 0; p < 5; ++p) {
        ABfrag B;  // lane holds W col (c0+ln16), k in table order
        const __bf16* wp = wcat + ((size_t)(p * 128 + k0 + kcb) * 128 + c0 + ln16);
#pragma unroll
        for (int s = 0; s < 16; ++s) B.v[s] = wp[(size_t)((s & 7) + (s >> 3) * 16) * 128];
        acc[p] = __builtin_amdgcn_wmma_f32_16x16x32_bf16(
            false, A.v, false, B.v, (short)0, acc[p], false, false);
      }
      if (ks < 3) A = An;
    }

    // ---- epilogue: mask * sigmoid gating; scatter right/gate via LDS
    {
      bf16x8 Lout;
#pragma unroll
      for (int v = 0; v < 8; ++v) {
        const int m = v + 8 * half;   // b_local (C-layout: VGPR v -> row m)
        const int b = b0 + m;
        const float mval = pmask[(size_t)a * NN + b];
        const float l_ = (acc[0][v] + vbL) * mval * sigm(acc[2][v] + vbLG);
        const float r_ = (acc[1][v] + vbR) * mval * sigm(acc[3][v] + vbRG);
        const float g_ = sigm(acc[4][v] + vbG);
        Lout[v] = (__bf16)l_;
        rstage[c * 136 + m * 8 + rt] = (__bf16)r_;
        gstage[m * 128 + c] = (__bf16)g_;
      }
      // left: per-lane 16B contiguous chunk in [c][a][b]
      *(bf16x8*)&leftT[(size_t)c * (NN * (size_t)NN) + (size_t)a * NN + (b0 + 8 * half)] = Lout;
    }
    __syncthreads();
    // ---- gate write-out: 16 rows x 128 ch bf16 = 4KB contiguous
    {
      const uint4* src = (const uint4*)gstage;
      uint4* dst = (uint4*)(gateO + ((size_t)a * NN + b0) * CP);
      dst[t] = src[t];
    }
  }
  __syncthreads();
  // ---- right write-out: [c][b][a] 16B chunks (8 'a' values each)
#pragma unroll
  for (int it = 0; it < 8; ++it) {
    const int u = t + it * 256;   // 0..2047
    const int cc = u >> 4;
    const int bb = u & 15;
    *(uint4*)&rightT[(size_t)cc * (NN * (size_t)NN) + (size_t)(b0 + bb) * NN + a0] =
        *(const uint4*)&rstage[cc * 136 + bb * 8];
  }
}

// ---------------------------------------------------------------- k_tri
// Per-channel GEMM: act[c][i][j] = sum_k L[c][i][k] * R[c][k][j].
// Block: 128x128 output tile for one channel plane; 8 waves in 2x4, each wave
// 64x32 (8 WMMA tiles). K streamed in 32-wide bf16 slabs, double-buffered in
// LDS and filled by CDNA5 async-to-LDS copies (4 per wave per slab) so the
// HBM stream overlaps the WMMA pipeline. s_wait_asynccnt 4 waits only for the
// older slab's copies (async loads complete in order). The k-loop is unrolled
// by 2 so the double-buffer index is compile-time constant.
__global__ __launch_bounds__(256) void k_tri(
    const __bf16* __restrict__ leftT, const __bf16* __restrict__ rightT,
    __bf16* __restrict__ actB) {
  __shared__ __bf16 Ls[2][128 * 40];   // [row i][k 0..31], pitch 40
  __shared__ __bf16 Rs[2][32 * 136];   // [row k][j 0..127], pitch 136

  const int t = threadIdx.x;
  const int lane = t & 31;
  const int w = t >> 5;
  const int wi = w >> 2;            // 0..1
  const int wj = w & 3;             // 0..3
  const int half = lane >> 4;
  const int ln16 = lane & 15;

  const int cch = blockIdx.z;
  const int i0 = blockIdx.y * 128;
  const int j0 = blockIdx.x * 128;
  const size_t plane = (size_t)cch * NN * NN;

  const __bf16* Lg = leftT + plane + (size_t)i0 * NN;   // rows i, cols k
  const __bf16* Rg = rightT + plane + j0;               // rows k, cols j

  const unsigned lsb0 = (unsigned)(size_t)&Ls[0][0];
  const unsigned lsb1 = (unsigned)(size_t)&Ls[1][0];
  const unsigned rsb0 = (unsigned)(size_t)&Rs[0][0];
  const unsigned rsb1 = (unsigned)(size_t)&Rs[1][0];

  // per-lane chunk ids (16B chunks): 512 chunks per slab, 2 per lane per buffer
  const int uL[2] = {(w * 2 + 0) * 32 + lane, (w * 2 + 1) * 32 + lane};

  f32x8 acc[4][2];
#pragma unroll
  for (int x = 0; x < 4; ++x)
#pragma unroll
    for (int y = 0; y < 2; ++y) acc[x][y] = (f32x8)(0.0f);

  // prologue: fill buffer 0
#pragma unroll
  for (int q = 0; q < 2; ++q) {
    const int u = uL[q];
    async_b128(lsb0 + (unsigned)((u >> 2) * 40 + (u & 3) * 8) * 2,
               Lg + (size_t)(u >> 2) * NN + (u & 3) * 8);
  }
#pragma unroll
  for (int q = 0; q < 2; ++q) {
    const int u = uL[q];
    async_b128(rsb0 + (unsigned)((u >> 4) * 136 + (u & 15) * 8) * 2,
               Rg + (size_t)(u >> 4) * NN + (u & 15) * 8);
  }

#pragma unroll 2
  for (int ks = 0; ks < 16; ++ks) {
    const int buf = ks & 1;   // compile-time under unroll-by-2
    const unsigned lsbN = buf ? lsb0 : lsb1;
    const unsigned rsbN = buf ? rsb0 : rsb1;
    if (ks < 15) {
      const int kn = (ks + 1) * 32;
#pragma unroll
      for (int q = 0; q < 2; ++q) {
        const int u = uL[q];
        async_b128(lsbN + (unsigned)((u >> 2) * 40 + (u & 3) * 8) * 2,
                   Lg + (size_t)(u >> 2) * NN + kn + (u & 3) * 8);
      }
#pragma unroll
      for (int q = 0; q < 2; ++q) {
        const int u = uL[q];
        async_b128(rsbN + (unsigned)((u >> 4) * 136 + (u & 15) * 8) * 2,
                   Rg + (size_t)(kn + (u >> 4)) * NN + (u & 15) * 8);
      }
      asm volatile("s_wait_asynccnt 0x4" ::: "memory");  // older slab done
    } else {
      asm volatile("s_wait_asynccnt 0x0" ::: "memory");
    }
    __syncthreads();

    ABfrag A[4], B[2];
#pragma unroll
    for (int ai = 0; ai < 4; ++ai) {
      const int row = wi * 64 + ai * 16 + ln16;
      const int kc = half ? 8 : 0;
      A[ai].h[0] = *(const bf16x8*)&Ls[buf][row * 40 + kc];
      A[ai].h[1] = *(const bf16x8*)&Ls[buf][row * 40 + kc + 16];
    }
#pragma unroll
    for (int bj = 0; bj < 2; ++bj) {
      const int jl = wj * 32 + bj * 16;
#pragma unroll
      for (int s = 0; s < 16; ++s)
        B[bj].v[s] = Rs[buf][(half * 8 + (s & 7) + (s >> 3) * 16) * 136 + jl + ln16];
    }
#pragma unroll
    for (int ai = 0; ai < 4; ++ai)
#pragma unroll
      for (int bj = 0; bj < 2; ++bj)
        acc[ai][bj] = __builtin_amdgcn_wmma_f32_16x16x32_bf16(
            false, A[ai].v, false, B[bj].v, (short)0, acc[ai][bj], false, false);
    __syncthreads();   // protect buffer reuse before next issue
  }

  // store bf16 [c][i][j] — lanes 0..15 contiguous in j -> 32B chunks
#pragma unroll
  for (int ai = 0; ai < 4; ++ai)
#pragma unroll
    for (int bj = 0; bj < 2; ++bj)
#pragma unroll
      for (int v = 0; v < 8; ++v) {
        const int ig = i0 + wi * 64 + ai * 16 + v + 8 * half;
        const int jg = j0 + wj * 32 + bj * 16 + ln16;
        actB[plane + (size_t)ig * NN + jg] = (__bf16)acc[ai][bj][v];
      }
}

// ---------------------------------------------------------------- k_out
// Tile 8x16 (i,j) x all 128 channels: load act[c][i][j] coalesced, transpose
// in LDS, LN over c, WMMA vs w_out, then (+b_out) * gate at the store.
__global__ __launch_bounds__(256) void k_out(
    const __bf16* __restrict__ actB,
    const float* __restrict__ lncw, const float* __restrict__ lncb,
    const __bf16* __restrict__ wcat,
    const float* __restrict__ bO,
    const __bf16* __restrict__ gateO,
    float* __restrict__ out) {
  __shared__ __bf16 trans[128 * 136];   // [ij 0..127][c], pitch 136
  __shared__ float stats[128 * 2];      // mean, rsig per ij row
  __shared__ float tstage[8][16 * 20];  // per-wave output tile [m][c_local], pitch 20

  const int t = threadIdx.x;
  const int lane = t & 31;
  const int w = t >> 5;
  const int half = lane >> 4;
  const int ln16 = lane & 15;

  const int i0 = blockIdx.y * 8;
  const int j0 = blockIdx.x * 16;

  // 1) load + transpose: [c][8i][16j] bf16 -> trans[ij][c]
#pragma unroll
  for (int it = 0; it < 8; ++it) {
    const int u = t + it * 256;     // 0..2047
    const int cc = u >> 4;
    const int ii = (u >> 1) & 7;
    const int sub = u & 1;
    const bf16x8 xv = *(const bf16x8*)&actB[(size_t)cc * NN * NN +
                                            (size_t)(i0 + ii) * NN + (j0 + sub * 8)];
#pragma unroll
    for (int jj = 0; jj < 8; ++jj)
      trans[(ii * 16 + sub * 8 + jj) * 136 + cc] = xv[jj];
  }
  __syncthreads();

  // 2) stats per ij row
  if (t < 128) {
    float sum = 0.f, sq = 0.f;
#pragma unroll 8
    for (int kk = 0; kk < 128; ++kk) {
      const float x = (float)trans[t * 136 + kk];
      sum += x; sq += x * x;
    }
    const float mean = sum * (1.0f / CP);
    const float var = sq * (1.0f / CP) - mean * mean;
    stats[t * 2] = mean;
    stats[t * 2 + 1] = rsqrtf(var + EPSV);
  }
  __syncthreads();

  // 3) normalize in place with ln_c scale/bias
  {
    const int row = t >> 1, hh = t & 1;
    const float mean = stats[row * 2], rs = stats[row * 2 + 1];
#pragma unroll
    for (int q = 0; q < 8; ++q) {
      const int ccb = hh * 64 + q * 8;
      bf16x8 xv = *(bf16x8*)&trans[row * 136 + ccb];
      bf16x8 yv;
#pragma unroll
      for (int e = 0; e < 8; ++e)
        yv[e] = (__bf16)(((float)xv[e] - mean) * rs * lncw[ccb + e] + lncb[ccb + e]);
      *(bf16x8*)&trans[row * 136 + ccb] = yv;
    }
  }
  __syncthreads();

  // 4) GEMM rows rt=w (16 ij rows per wave), all 8 output-channel tiles
  const __bf16* wout = wcat + 5 * 16384;
  for (int ct = 0; ct < 8; ++ct) {
    const int c0 = ct * 16;
    f32x8 acc = (f32x8)(0.0f);
#pragma unroll
    for (int ks = 0; ks < 4; ++ks) {
      const int k0 = ks * 32;
      ABfrag A, B;
      const int kc = k0 + (half ? 8 : 0);
      A.h[0] = *(const bf16x8*)&trans[(w * 16 + ln16) * 136 + kc];
      A.h[1] = *(const bf16x8*)&trans[(w * 16 + ln16) * 136 + kc + 16];
      const __bf16* wp = wout + (size_t)kc * 128 + c0 + ln16;
#pragma unroll
      for (int s = 0; s < 16; ++s) B.v[s] = wp[(size_t)((s & 7) + (s >> 3) * 16) * 128];
      acc = __builtin_amdgcn_wmma_f32_16x16x32_bf16(
          false, A.v, false, B.v, (short)0, acc, false, false);
    }
    const float bias = bO[c0 + ln16];
#pragma unroll
    for (int v = 0; v < 8; ++v)
      tstage[w][(v + 8 * half) * 20 + ln16] = acc[v] + bias;
    __syncthreads();   // uniform control flow: all waves run identical ct loop
#pragma unroll
    for (int it = 0; it < 2; ++it) {
      const int q = it * 32 + lane;            // 0..63 chunks of 16B
      const int m = q >> 2, sub = q & 3;
      const int r = w * 16 + m;                // ij local row
      const int ii = i0 + (r >> 4), jj = j0 + (r & 15);
      const size_t base = ((size_t)ii * NN + jj) * CP + c0 + sub * 4;
      float4 val = *(float4*)&tstage[w][m * 20 + sub * 4];
      const __bf16* gp = gateO + base;
      val.x *= (float)gp[0];
      val.y *= (float)gp[1];
      val.z *= (float)gp[2];
      val.w *= (float)gp[3];
      *(float4*)&out[base] = val;
    }
    __syncthreads();
  }
}

// ---------------------------------------------------------------- launch
extern "C" void kernel_launch(void* const* d_in, const int* in_sizes, int n_in,
                              void* d_out, int out_size, void* d_ws, size_t ws_size,
                              hipStream_t stream) {
  const float* pair_act  = (const float*)d_in[0];
  const float* pair_mask = (const float*)d_in[1];
  const float* ln_in_w   = (const float*)d_in[2];
  const float* ln_in_b   = (const float*)d_in[3];
  const float* w_left    = (const float*)d_in[4];
  const float* b_left    = (const float*)d_in[5];
  const float* w_right   = (const float*)d_in[6];
  const float* b_right   = (const float*)d_in[7];
  const float* w_lgate   = (const float*)d_in[8];
  const float* b_lgate   = (const float*)d_in[9];
  const float* w_rgate   = (const float*)d_in[10];
  const float* b_rgate   = (const float*)d_in[11];
  const float* ln_c_w    = (const float*)d_in[12];
  const float* ln_c_b    = (const float*)d_in[13];
  const float* w_out_p   = (const float*)d_in[14];
  const float* b_out_p   = (const float*)d_in[15];
  const float* w_gate    = (const float*)d_in[16];
  const float* b_gate    = (const float*)d_in[17];

  char* ws = (char*)d_ws;
  const size_t SZ = 67108864ull;  // 64 MiB per bf16 buffer
  __bf16* wcat   = (__bf16*)(ws);
  __bf16* leftT  = (__bf16*)(ws + (1ull << 20));
  __bf16* rightT = (__bf16*)(ws + (1ull << 20) + SZ);
  __bf16* gateO  = (__bf16*)(ws + (1ull << 20) + 2 * SZ);
  __bf16* actB   = (__bf16*)(ws + (1ull << 20) + 3 * SZ);

  k_convert<<<384, 256, 0, stream>>>(w_left, w_right, w_lgate, w_rgate,
                                     w_gate, w_out_p, wcat);

  k_proj<<<dim3(32, 64), 256, 0, stream>>>(
      pair_act, pair_mask, ln_in_w, ln_in_b,
      b_left, b_right, b_lgate, b_rgate, b_gate,
      wcat, leftT, rightT, gateO);

  k_tri<<<dim3(4, 4, 128), 256, 0, stream>>>(leftT, rightT, actB);

  k_out<<<dim3(32, 64), 256, 0, stream>>>(
      actB, ln_c_w, ln_c_b, wcat, b_out_p, gateO, (float*)d_out);
}
